// MoE_22110491640665
// MI455X (gfx1250) — matrix operand order repaired and checked by reading, not compiled, not run on hardware
//
#include <hip/hip_runtime.h>
#include <hip/hip_bf16.h>

#define B_TOK 4096
#define D_DIM 512
#define H_DIM 1024
#define O_DIM 512
#define E_NUM 8
#define K_TOP 4
#define BT    32          // token tile per block

typedef __attribute__((ext_vector_type(16))) __bf16 v16bf;
typedef __attribute__((ext_vector_type(8)))  float  v8f;

union Frag { unsigned int u[8]; v16bf v; };

static __device__ __forceinline__ unsigned short f2bf(float f) {
  unsigned int u = __float_as_uint(f);
  u += 0x7FFFu + ((u >> 16) & 1u);   // round-to-nearest-even
  return (unsigned short)(u >> 16);
}

#if __has_builtin(__builtin_amdgcn_cvt_pk_bf16_f32)
typedef __attribute__((ext_vector_type(2))) __bf16 v2bf;
static __device__ __forceinline__ unsigned int f2bf_pk(float a, float b) {
  union { v2bf v; unsigned int u; } cv;
  cv.v = __builtin_amdgcn_cvt_pk_bf16_f32(a, b);
  return cv.u;
}
#else
static __device__ __forceinline__ unsigned int f2bf_pk(float a, float b) {
  return (unsigned int)f2bf(a) | ((unsigned int)f2bf(b) << 16);
}
#endif

// CDNA5 async global->LDS copy (ASYNCcnt path), CDNA5 ISA
static __device__ __forceinline__ void async_copy_b128(void* lds_dst,
                                                       const void* gsrc) {
  unsigned lds_off = (unsigned)(size_t)lds_dst;   // LDS offset = addr[31:0]
  asm volatile("global_load_async_to_lds_b128 %0, %1, off"
               :: "v"(lds_off), "v"(gsrc) : "memory");
}
static __device__ __forceinline__ void wait_asynccnt0() {
  asm volatile("s_wait_asynccnt 0" ::: "memory");
}

// ---------------------------------------------------------------------------
// 1) Gating: logits = x @ w_gate ; top-4 ; softmax -> dense gates (B x E)
// ---------------------------------------------------------------------------
__global__ __launch_bounds__(256) void moe_gate_kernel(
    const float* __restrict__ x, const float* __restrict__ wg,
    float* __restrict__ gates) {
  __shared__ float swg[D_DIM * E_NUM];   // 16 KB
  const int tid = threadIdx.x;
  for (int i = tid; i < D_DIM * E_NUM; i += 256) swg[i] = wg[i];
  __syncthreads();

  const int b = blockIdx.x * 256 + tid;
  const float* xr = x + (size_t)b * D_DIM;
  float acc[E_NUM];
#pragma unroll
  for (int e = 0; e < E_NUM; ++e) acc[e] = 0.f;
  for (int d = 0; d < D_DIM; ++d) {
    float xv = xr[d];
#pragma unroll
    for (int e = 0; e < E_NUM; ++e) acc[e] += xv * swg[d * E_NUM + e];
  }
  float vals[E_NUM];
#pragma unroll
  for (int e = 0; e < E_NUM; ++e) vals[e] = acc[e];
  int   idx[K_TOP];
  float tv[K_TOP];
#pragma unroll
  for (int k = 0; k < K_TOP; ++k) {
    int   bi = 0;
    float bv = vals[0];
#pragma unroll
    for (int e = 1; e < E_NUM; ++e)
      if (vals[e] > bv) { bv = vals[e]; bi = e; }
    idx[k] = bi; tv[k] = bv; vals[bi] = -3.4e38f;
  }
  float mx = tv[0], s = 0.f, ex[K_TOP];
#pragma unroll
  for (int k = 0; k < K_TOP; ++k) { ex[k] = __expf(tv[k] - mx); s += ex[k]; }
  float g[E_NUM];
#pragma unroll
  for (int e = 0; e < E_NUM; ++e) g[e] = 0.f;
#pragma unroll
  for (int k = 0; k < K_TOP; ++k) g[idx[k]] = ex[k] / s;
#pragma unroll
  for (int e = 0; e < E_NUM; ++e) gates[(size_t)b * E_NUM + e] = g[e];
}

// ---------------------------------------------------------------------------
// 2) Load-balance loss (deterministic reduction, no atomics)
// ---------------------------------------------------------------------------
__global__ __launch_bounds__(256) void moe_loss_kernel(
    const float* __restrict__ gates, float* __restrict__ out_loss) {
  __shared__ float s_imp[E_NUM][256];
  __shared__ float s_load[E_NUM][256];
  const int tid = threadIdx.x;
  float imp[E_NUM], ld[E_NUM];
#pragma unroll
  for (int e = 0; e < E_NUM; ++e) { imp[e] = 0.f; ld[e] = 0.f; }
  for (int b = tid; b < B_TOK; b += 256) {
#pragma unroll
    for (int e = 0; e < E_NUM; ++e) {
      float g = gates[(size_t)b * E_NUM + e];
      imp[e] += g;
      ld[e]  += (g > 0.f) ? 1.f : 0.f;
    }
  }
#pragma unroll
  for (int e = 0; e < E_NUM; ++e) { s_imp[e][tid] = imp[e]; s_load[e][tid] = ld[e]; }
  __syncthreads();
  if (tid == 0) {
    float loss = 0.f;
    for (int which = 0; which < 2; ++which) {
      float v[E_NUM];
      for (int e = 0; e < E_NUM; ++e) {
        float s = 0.f;
        for (int t = 0; t < 256; ++t)
          s += (which == 0) ? s_imp[e][t] : s_load[e][t];
        v[e] = s;
      }
      float mean = 0.f;
      for (int e = 0; e < E_NUM; ++e) mean += v[e];
      mean /= (float)E_NUM;
      float var = 0.f;
      for (int e = 0; e < E_NUM; ++e) { float d = v[e] - mean; var += d * d; }
      var /= (float)(E_NUM - 1);             // ddof = 1
      loss += var / (mean * mean + 1e-10f);
    }
    *out_loss = loss * 0.01f;
  }
}

// ---------------------------------------------------------------------------
// 3) bf16 conversion / transposition
// ---------------------------------------------------------------------------
__global__ void conv_x_kernel(const float* __restrict__ x,
                              unsigned int* __restrict__ xb) {
  int i = blockIdx.x * 256 + threadIdx.x;   // 8 floats -> 4 packed pairs
  const float4* src = (const float4*)(x + (size_t)i * 8);
  float4 a = src[0], b = src[1];
  uint4 o;
  o.x = f2bf_pk(a.x, a.y);
  o.y = f2bf_pk(a.z, a.w);
  o.z = f2bf_pk(b.x, b.y);
  o.w = f2bf_pk(b.z, b.w);
  ((uint4*)xb)[i] = o;
}

// Transpose+convert: in[e][RIN][CIN] f32 -> out[e][CIN][RIN] bf16
// 64x64 tiles via LDS, coalesced both ways. 256 threads, 16 elems/thread.
template <int RIN, int CIN>
__global__ __launch_bounds__(256) void conv_transpose_kernel(
    const float* __restrict__ in, unsigned short* __restrict__ out) {
  __shared__ unsigned short tile[64][66];   // padded, b32-aligned pairs
  const int ntc = CIN / 64, ntr = RIN / 64;
  int bx = blockIdx.x;
  int e  = bx / (ntr * ntc);
  int r  = bx % (ntr * ntc);
  int r0 = (r / ntc) * 64, c0 = (r % ntc) * 64;

  int base = threadIdx.x * 16;
  int rr = base >> 6, cc = base & 63;       // 16 contiguous input cols
  const float* src = in + ((size_t)e * RIN + r0 + rr) * CIN + c0 + cc;
#pragma unroll
  for (int k = 0; k < 16; k += 2)
    *(unsigned int*)&tile[rr][cc + k] = f2bf_pk(src[k], src[k + 1]);
  __syncthreads();

  int oc = base >> 6, od = base & 63;       // 16 contiguous output cols (RIN dim)
  unsigned short* dst = out + ((size_t)e * CIN + c0 + oc) * RIN + r0 + od;
#pragma unroll
  for (int k = 0; k < 16; k += 2) {
    unsigned int p = (unsigned int)tile[od + k][oc] |
                     ((unsigned int)tile[od + k + 1][oc] << 16);
    *(unsigned int*)&dst[k] = p;
  }
}

// ---------------------------------------------------------------------------
// 4) Fused experts: 32-token tile per block.
//    Wave w -> row-half rh = w>>2 (16 rows), col-quarter cq = w&3.
//    GEMM1: x(LDS, async-staged) @ W1t -> relu -> h_lds (bf16)
//    GEMM2: h_lds @ W2t -> softmax (shfl + LDS combine) -> y regs
// ---------------------------------------------------------------------------
__global__ __launch_bounds__(256) void moe_fused_kernel(
    const unsigned short* __restrict__ xbf,
    const unsigned short* __restrict__ w1t,
    const unsigned short* __restrict__ w2t,
    const float* __restrict__ b1, const float* __restrict__ b2,
    const float* __restrict__ gates, float* __restrict__ y) {
  __shared__ __align__(16) unsigned short h_lds[BT * H_DIM]; // 64 KB
  __shared__ __align__(16) unsigned short x_lds[BT * D_DIM]; // 32 KB
  __shared__ float red_max[BT][4];
  __shared__ float red_sum[BT][4];

  const int tid  = threadIdx.x;
  const int w    = tid >> 5;      // wave 0..7
  const int lane = tid & 31;
  const int lm   = lane & 15;
  const int lh   = lane >> 4;
  const int rh   = w >> 2;        // row-half: rows [rh*16, rh*16+16)
  const int cq   = w & 3;         // col quarter
  const int b0   = blockIdx.x * BT;
  const int row16 = rh * 16;      // wave's first row within tile

  // ---- stage x tile (BT x D, bf16, 32 KB) into LDS via async copies ----
  {
    const char* gsrc = (const char*)(xbf + (size_t)b0 * D_DIM);
    char*       ldst = (char*)x_lds;
#pragma unroll
    for (int i = 0; i < (BT * D_DIM * 2) / (256 * 16); ++i) {
      int off = (tid + i * 256) * 16;
      async_copy_b128(ldst + off, gsrc + off);
    }
    wait_asynccnt0();
  }
  __syncthreads();

  v8f yacc[8];                    // 16 rows x 128 cols per wave
#pragma unroll
  for (int t = 0; t < 8; ++t)
#pragma unroll
    for (int j = 0; j < 8; ++j) yacc[t][j] = 0.f;

  for (int e = 0; e < E_NUM; ++e) {
    // ---------- GEMM1: h(16 x 256 cols) in two 128-col passes ----------
#pragma unroll
    for (int pass = 0; pass < 2; ++pass) {
      v8f hacc[8];
#pragma unroll
      for (int t = 0; t < 8; ++t)
#pragma unroll
        for (int j = 0; j < 8; ++j) hacc[t][j] = 0.f;

      for (int ks = 0; ks < D_DIM; ks += 32) {
        Frag fa;                                 // A: x tile from LDS
#pragma unroll
        for (int v = 0; v < 8; ++v) {
          int k = ks + 16 * lh + 2 * v;
          fa.u[v] = *(const unsigned int*)(x_lds + (row16 + lm) * D_DIM + k);
        }
#pragma unroll
        for (int t = 0; t < 8; ++t) {
          int hcol = cq * 256 + pass * 128 + t * 16 + lm;
          const unsigned short* wc = w1t + ((size_t)e * H_DIM + hcol) * D_DIM;
          Frag fb;                               // B: W1t 32x16 bf16
#pragma unroll
          for (int v = 0; v < 8; ++v) {
            int k = ks + 16 * lh + 2 * v;
            fb.u[v] = *(const unsigned int*)(wc + k);
          }
          hacc[t] = __builtin_amdgcn_wmma_f32_16x16x32_bf16(
              false, fa.v, false, fb.v, (short)0, hacc[t], false, false);
        }
      }
      // bias + relu -> bf16 h tile in LDS (row-major, K contiguous)
#pragma unroll
      for (int t = 0; t < 8; ++t) {
        int hcol = cq * 256 + pass * 128 + t * 16 + lm;
        float bias = b1[e * H_DIM + hcol];
#pragma unroll
        for (int j = 0; j < 8; j += 2) {
          int row = row16 + j + 8 * lh;
          float v0 = fmaxf(hacc[t][j]     + bias, 0.f);
          float v1 = fmaxf(hacc[t][j + 1] + bias, 0.f);
          unsigned int p = f2bf_pk(v0, v1);
          h_lds[row * H_DIM + hcol]       = (unsigned short)p;
          h_lds[(row + 1) * H_DIM + hcol] = (unsigned short)(p >> 16);
        }
      }
    }
    __syncthreads();

    // ---------- GEMM2: out(16 x 128) cols [cq*128, cq*128+128) ----------
    v8f oacc[8];
#pragma unroll
    for (int t = 0; t < 8; ++t)
#pragma unroll
      for (int j = 0; j < 8; ++j) oacc[t][j] = 0.f;

    for (int ks = 0; ks < H_DIM; ks += 32) {
      Frag fa;                                   // A: h tile from LDS
#pragma unroll
      for (int v = 0; v < 8; ++v) {
        int k = ks + 16 * lh + 2 * v;
        fa.u[v] = *(const unsigned int*)(h_lds + (row16 + lm) * H_DIM + k);
      }
#pragma unroll
      for (int t = 0; t < 8; ++t) {
        int ocol = cq * 128 + t * 16 + lm;
        const unsigned short* wc = w2t + ((size_t)e * O_DIM + ocol) * H_DIM;
        Frag fb;                                 // B: W2t 32x16 bf16
#pragma unroll
        for (int v = 0; v < 8; ++v) {
          int k = ks + 16 * lh + 2 * v;
          fb.u[v] = *(const unsigned int*)(wc + k);
        }
        oacc[t] = __builtin_amdgcn_wmma_f32_16x16x32_bf16(
            false, fa.v, false, fb.v, (short)0, oacc[t], false, false);
      }
    }
    // add bias in place
#pragma unroll
    for (int t = 0; t < 8; ++t) {
      float bias = b2[e * O_DIM + cq * 128 + t * 16 + lm];
#pragma unroll
      for (int j = 0; j < 8; ++j) oacc[t][j] += bias;
    }

    // ---------- row softmax over O=512 (4 waves per row-half) ----------
    float pm[8];
#pragma unroll
    for (int j = 0; j < 8; ++j) {
      float m = oacc[0][j];
#pragma unroll
      for (int t = 1; t < 8; ++t) m = fmaxf(m, oacc[t][j]);
#pragma unroll
      for (int off = 1; off < 16; off <<= 1)
        m = fmaxf(m, __shfl_xor(m, off, 32));
      pm[j] = m;
    }
    if (lm == 0) {
#pragma unroll
      for (int j = 0; j < 8; ++j) red_max[row16 + j + 8 * lh][cq] = pm[j];
    }
    __syncthreads();
    float rmax[8];
#pragma unroll
    for (int j = 0; j < 8; ++j) {
      int r = row16 + j + 8 * lh;
      float m = red_max[r][0];
#pragma unroll
      for (int w2 = 1; w2 < 4; ++w2) m = fmaxf(m, red_max[r][w2]);
      rmax[j] = m;
    }
    float ps[8];
#pragma unroll
    for (int j = 0; j < 8; ++j) ps[j] = 0.f;
#pragma unroll
    for (int t = 0; t < 8; ++t)
#pragma unroll
      for (int j = 0; j < 8; ++j) {
        oacc[t][j] = __expf(oacc[t][j] - rmax[j]);
        ps[j] += oacc[t][j];
      }
#pragma unroll
    for (int j = 0; j < 8; ++j) {
#pragma unroll
      for (int off = 1; off < 16; off <<= 1)
        ps[j] += __shfl_xor(ps[j], off, 32);
    }
    if (lm == 0) {
#pragma unroll
      for (int j = 0; j < 8; ++j) red_sum[row16 + j + 8 * lh][cq] = ps[j];
    }
    __syncthreads();
#pragma unroll
    for (int j = 0; j < 8; ++j) {
      int r = row16 + j + 8 * lh;
      float s = red_sum[r][0];
#pragma unroll
      for (int w2 = 1; w2 < 4; ++w2) s += red_sum[r][w2];
      float g = gates[(size_t)(b0 + r) * E_NUM + e];
      float scale = g / s;
#pragma unroll
      for (int t = 0; t < 8; ++t) yacc[t][j] += oacc[t][j] * scale;
    }
    __syncthreads();   // protect h_lds / red buffers for next expert
  }

  // ---------- write y ----------
#pragma unroll
  for (int t = 0; t < 8; ++t)
#pragma unroll
    for (int j = 0; j < 8; ++j)
      y[(size_t)(b0 + row16 + j + 8 * lh) * O_DIM + cq * 128 + t * 16 + lm] =
          yacc[t][j];
}

// ---------------------------------------------------------------------------
extern "C" void kernel_launch(void* const* d_in, const int* in_sizes, int n_in,
                              void* d_out, int out_size, void* d_ws,
                              size_t ws_size, hipStream_t stream) {
  const float* x  = (const float*)d_in[0];   // (4096, 512)
  const float* wg = (const float*)d_in[1];   // (512, 8)
  const float* W1 = (const float*)d_in[2];   // (8, 512, 1024)
  const float* b1 = (const float*)d_in[3];   // (8, 1024)
  const float* W2 = (const float*)d_in[4];   // (8, 1024, 512)
  const float* b2 = (const float*)d_in[5];   // (8, 512)

  float* out  = (float*)d_out;               // y (4096x512) then loss scalar
  float* y    = out;
  float* loss = out + (size_t)B_TOK * O_DIM;

  char* ws = (char*)d_ws;
  float*          gates = (float*)ws;                         // 128 KB
  unsigned int*   xbf   = (unsigned int*)(ws + (131072));     // 4 MB (bf16 x2)
  unsigned short* w1t   = (unsigned short*)(ws + (131072 + 4194304));
  unsigned short* w2t   = (unsigned short*)(ws + (131072 + 4194304 + 8388608));

  moe_gate_kernel<<<B_TOK / 256, 256, 0, stream>>>(x, wg, gates);
  moe_loss_kernel<<<1, 256, 0, stream>>>(gates, loss);

  conv_x_kernel<<<(B_TOK * D_DIM) / (256 * 8), 256, 0, stream>>>(x, xbf);
  conv_transpose_kernel<D_DIM, H_DIM>
      <<<E_NUM * (D_DIM / 64) * (H_DIM / 64), 256, 0, stream>>>(W1, w1t);
  conv_transpose_kernel<H_DIM, O_DIM>
      <<<E_NUM * (H_DIM / 64) * (O_DIM / 64), 256, 0, stream>>>(W2, w2t);

  moe_fused_kernel<<<B_TOK / BT, 256, 0, stream>>>(
      (const unsigned short*)xbf, w1t, w2t, b1, b2, gates, y);
}
